// CrossPatchSelfAttention_2920577762065
// MI455X (gfx1250) — matrix operand order repaired and checked by reading, not compiled, not run on hardware
//
#include <hip/hip_runtime.h>
#include <cstdint>
#include <cstddef>

// ---------------------------------------------------------------------------
// Transformer block on gfx1250 (MI455X): bf16 WMMA for all GEMM-shaped math,
// flash attention, async global->LDS double-buffered GEMM pipeline,
// permlane16 VALU softmax reductions (no LDS round-trips).
// ---------------------------------------------------------------------------

typedef __attribute__((ext_vector_type(16))) __bf16 bf16x16;
typedef __attribute__((ext_vector_type(4)))  __bf16 bf16x4;
typedef __attribute__((ext_vector_type(8)))  float  f32x8;
typedef __attribute__((ext_vector_type(4)))  int    v4i;

#define WMMA_BF16(a, b, c) \
  __builtin_amdgcn_wmma_f32_16x16x32_bf16(false, (a), false, (b), (short)0, (c), false, false)

// ---- CDNA5 async global->LDS copy (ASYNCcnt path), with fallback ------------
#if defined(__HIP_DEVICE_COMPILE__) && \
    __has_builtin(__builtin_amdgcn_global_load_async_to_lds_b128)
#define HAS_ASYNC 1
#else
#define HAS_ASYNC 0
#endif

#if HAS_ASYNC
typedef __attribute__((address_space(1))) v4i v4i_g;
typedef __attribute__((address_space(3))) v4i v4i_l;
__device__ __forceinline__ void async_copy16(const __bf16* g, __bf16* l) {
  __builtin_amdgcn_global_load_async_to_lds_b128(
      (v4i_g*)(uintptr_t)g, (v4i_l*)(uintptr_t)l, 0, 0);
}
#define WAIT_ASYNC(n) asm volatile("s_wait_asynccnt " #n ::: "memory")
#else
#define WAIT_ASYNC(n)
#endif

// ---- 16-lane XOR butterfly: VALU permlane16 when available, else shfl -------
#if defined(__HIP_DEVICE_COMPILE__) && __has_builtin(__builtin_amdgcn_permlane16)
#define HAS_PERMLANE16 1
#else
#define HAS_PERMLANE16 0
#endif

template <int OFF>
__device__ __forceinline__ float xor16(float v) {
#if HAS_PERMLANE16
  // per-row nibble selects for lane i <- lane (i ^ OFF), rows of 16 lanes
  int sel0, sel1;
  if (OFF == 1)      { sel0 = (int)0x67452301u; sel1 = (int)0xEFCDAB89u; }
  else if (OFF == 2) { sel0 = (int)0x54761032u; sel1 = (int)0xDCFE98BAu; }
  else if (OFF == 4) { sel0 = (int)0x32107654u; sel1 = (int)0xBA98FEDCu; }
  else               { sel0 = (int)0xFEDCBA98u; sel1 = (int)0x76543210u; }
  int i = __float_as_int(v);
  return __int_as_float(__builtin_amdgcn_permlane16(i, i, sel0, sel1, false, false));
#else
  return __shfl_xor(v, OFF, 32);
#endif
}

__device__ __forceinline__ float redmax16(float v) {
  v = fmaxf(v, xor16<1>(v));
  v = fmaxf(v, xor16<2>(v));
  v = fmaxf(v, xor16<4>(v));
  v = fmaxf(v, xor16<8>(v));
  return v;
}
__device__ __forceinline__ float redsum16(float v) {
  v += xor16<1>(v);
  v += xor16<2>(v);
  v += xor16<4>(v);
  v += xor16<8>(v);
  return v;
}

// ---- fragment loaders -------------------------------------------------------
// A-matrix 16x32 bf16 fragment (ISA 7.12.2): lane L holds row M = L%16;
// lanes<16: K {0..7} and {16..23}; lanes>=16: K {8..15} and {24..31}.
__device__ __forceinline__ bf16x16 frag_a(const __bf16* base, int row0, int ldk,
                                          int kofs, int lane) {
  int r  = row0 + (lane & 15);
  int kb = kofs + ((lane & 16) ? 8 : 0);
  const uint4* p = (const uint4*)(base + (size_t)r * ldk + kb);
  union { bf16x16 v; uint4 u[2]; } t;
  t.u[0] = p[0];
  t.u[1] = p[2];            // +16 elements (= +32 bytes)
  return t.v;
}

// B-matrix 32x16 bf16 fragment: lane L holds column N = L%16;
// lanes<16: K {0..7} and {16..23}; lanes>=16: K {8..15} and {24..31}.
// "base" is B^T stored row-major [N][K] -> contiguous in K per column.
__device__ __forceinline__ bf16x16 frag_b(const __bf16* base, int n0, int ldk,
                                          int kofs, int lane) {
  int r  = n0 + (lane & 15);
  int kb = kofs + ((lane & 16) ? 8 : 0);
  const uint4* p = (const uint4*)(base + (size_t)r * ldk + kb);
  union { bf16x16 v; uint4 u[2]; } t;
  t.u[0] = p[0];
  t.u[1] = p[2];
  return t.v;
}

// ---- LayerNorm (fp32 in -> bf16 out), one 256-thread block per row ----------
__global__ __launch_bounds__(256) void ln_to_bf16(const float* __restrict__ x,
                                                  const float* __restrict__ g,
                                                  const float* __restrict__ bta,
                                                  __bf16* __restrict__ out) {
  int row  = blockIdx.x;
  int tid  = threadIdx.x;
  int wave = tid >> 5, lane = tid & 31;
  const float4* xr = (const float4*)(x + (size_t)row * 1024);
  float4 v = xr[tid];
  float s  = v.x + v.y + v.z + v.w;
  float s2 = v.x * v.x + v.y * v.y + v.z * v.z + v.w * v.w;
#pragma unroll
  for (int off = 16; off > 0; off >>= 1) {
    s  += __shfl_xor(s, off, 32);
    s2 += __shfl_xor(s2, off, 32);
  }
  __shared__ float rs[8], rs2[8];
  if (lane == 0) { rs[wave] = s; rs2[wave] = s2; }
  __syncthreads();
  float ts = 0.f, ts2 = 0.f;
#pragma unroll
  for (int i = 0; i < 8; ++i) { ts += rs[i]; ts2 += rs2[i]; }
  float mu   = ts * (1.0f / 1024.0f);
  float var  = ts2 * (1.0f / 1024.0f) - mu * mu;
  float rstd = rsqrtf(var + 1e-5f);
  float4 gg = ((const float4*)g)[tid];
  float4 bb = ((const float4*)bta)[tid];
  bf16x4 o;
  o[0] = (__bf16)((v.x - mu) * rstd * gg.x + bb.x);
  o[1] = (__bf16)((v.y - mu) * rstd * gg.y + bb.y);
  o[2] = (__bf16)((v.z - mu) * rstd * gg.z + bb.z);
  o[3] = (__bf16)((v.w - mu) * rstd * gg.w + bb.w);
  ((bf16x4*)(out + (size_t)row * 1024))[tid] = o;
}

// ---- weight transpose+convert: in fp32 [K][N] -> out bf16 [N][K] ------------
__global__ __launch_bounds__(256) void transpose_to_bf16(const float* __restrict__ in,
                                                         __bf16* __restrict__ out,
                                                         int K, int N) {
  __shared__ float t[32][33];
  int n0 = blockIdx.x * 32, k0 = blockIdx.y * 32;
  int tx = threadIdx.x, ty = threadIdx.y;   // 32 x 8
#pragma unroll
  for (int i = 0; i < 32; i += 8)
    t[ty + i][tx] = in[(size_t)(k0 + ty + i) * N + n0 + tx];
  __syncthreads();
#pragma unroll
  for (int i = 0; i < 32; i += 8)
    out[(size_t)(n0 + ty + i) * K + k0 + tx] = (__bf16)t[tx][ty + i];
}

// ---- tiled WMMA GEMM:  C[m][n] = sum_k A[m][k] * Bt[n][k] + bias[n] ---------
// Block tile 128x128, K-step 64, double-buffered async global->LDS pipeline.
// 8 waves -> each wave 32x64 (2x4 WMMA tiles), 16 WMMA per K-step.
enum { EP_BF16 = 0, EP_GELU = 1, EP_RESID = 2, EP_QKV = 3 };
#define LDT 72   // 64 + 8 bf16 pad: conflict-free 16B fragment loads

template <int MODE>
__global__ __launch_bounds__(256) void gemm_nt(
    const __bf16* __restrict__ A, const __bf16* __restrict__ Bt,
    const float* __restrict__ bias, __bf16* __restrict__ Cbf,
    float* __restrict__ C32, const float* __restrict__ resid,
    __bf16* __restrict__ qo, __bf16* __restrict__ ko, __bf16* __restrict__ vo,
    int M, int N, int K) {
  __shared__ __bf16 As[2][128 * LDT];
  __shared__ __bf16 Bs[2][128 * LDT];

  int tid = threadIdx.x, wave = tid >> 5, lane = tid & 31;
  int wm = wave & 3, wn = wave >> 2;          // 4 (M) x 2 (N) waves
  int bm = blockIdx.y * 128, bn = blockIdx.x * 128;

  f32x8 acc[2][4] = {};

  // tile-load mapping: thread -> (row, 32-elem chunk): 64B each matrix
  int r = tid >> 1, c = (tid & 1) * 32;
  const __bf16* gA = A + (size_t)(bm + r) * K + c;
  const __bf16* gB = Bt + (size_t)(bn + r) * K + c;

  auto issue_stage = [&](int kt, int buf) {
    const __bf16* pa = gA + (size_t)kt * 64;
    const __bf16* pb = gB + (size_t)kt * 64;
    __bf16* la = &As[buf][r * LDT + c];
    __bf16* lb = &Bs[buf][r * LDT + c];
#if HAS_ASYNC
#pragma unroll
    for (int i = 0; i < 4; ++i) {
      async_copy16(pa + i * 8, la + i * 8);
      async_copy16(pb + i * 8, lb + i * 8);
    }
#else
    const uint4* qa = (const uint4*)pa;
    const uint4* qb = (const uint4*)pb;
    uint4* sa = (uint4*)la;
    uint4* sb = (uint4*)lb;
#pragma unroll
    for (int i = 0; i < 4; ++i) { sa[i] = qa[i]; sb[i] = qb[i]; }
#endif
  };

  const int kTiles = K >> 6;
  issue_stage(0, 0);
  for (int kt = 0; kt < kTiles; ++kt) {
    int buf = kt & 1;
    if (kt + 1 < kTiles) {
      issue_stage(kt + 1, buf ^ 1);   // prefetch next stage (8 async ops/wave)
      WAIT_ASYNC(0x8);                // previous stage complete
    } else {
      WAIT_ASYNC(0x0);
    }
    __syncthreads();
#pragma unroll
    for (int kc = 0; kc < 64; kc += 32) {
      bf16x16 af[2], bfr[4];
      af[0] = frag_a(&As[buf][0], wm * 32,      LDT, kc, lane);
      af[1] = frag_a(&As[buf][0], wm * 32 + 16, LDT, kc, lane);
#pragma unroll
      for (int nt = 0; nt < 4; ++nt)
        bfr[nt] = frag_b(&Bs[buf][0], wn * 64 + nt * 16, LDT, kc, lane);
#pragma unroll
      for (int mt = 0; mt < 2; ++mt)
#pragma unroll
        for (int nt = 0; nt < 4; ++nt)
          acc[mt][nt] = WMMA_BF16(af[mt], bfr[nt], acc[mt][nt]);
    }
    __syncthreads();   // all waves done with buf before it is overwritten
  }

  // ---- epilogue (C-layout: VGPR j -> row j (lanes<16) / row j+8 (lanes>=16),
  //                column = lane % 16) ----
  int lane15 = lane & 15;
  int rsel   = (lane & 16) ? 8 : 0;
#pragma unroll
  for (int mt = 0; mt < 2; ++mt) {
#pragma unroll
    for (int nt = 0; nt < 4; ++nt) {
      int col  = bn + wn * 64 + nt * 16 + lane15;
      float bv = bias[col];
#pragma unroll
      for (int j = 0; j < 8; ++j) {
        int row = bm + wm * 32 + mt * 16 + rsel + j;
        float v = acc[mt][nt][j] + bv;
        if (MODE == EP_BF16) {
          Cbf[(size_t)row * N + col] = (__bf16)v;
        } else if (MODE == EP_GELU) {
          float gel = 0.5f * v * (1.0f + erff(v * 0.70710678f));
          Cbf[(size_t)row * N + col] = (__bf16)gel;
        } else if (MODE == EP_RESID) {
          C32[(size_t)row * N + col] = resid[(size_t)row * N + col] + v;
        } else {  // EP_QKV: scatter to q/k [B,H,N,64] and v^T [B,H,64,N]
          int sec = col >> 10;
          int din = col & 1023;
          int hh = din >> 6, hd = din & 63;
          int bi = row >> 10, ni = row & 1023;
          size_t bh = (size_t)bi * 16 + hh;
          if (sec == 0)      qo[(bh * 1024 + ni) * 64 + hd] = (__bf16)v;
          else if (sec == 1) ko[(bh * 1024 + ni) * 64 + hd] = (__bf16)v;
          else               vo[(bh * 64 + hd) * 1024 + ni] = (__bf16)v;
        }
      }
    }
  }
}

// ---- flash attention: per block one (b,h) x 128 queries; wave = 16 queries --
// The P tile is wave-private and DS ops from one wave are processed in order
// (ISA 7.3), so the key loop needs no workgroup barriers at all -- only
// compiler-level fences to preserve the DS program order.
#define PLD 72   // 64 + 8 pad for per-wave P tile
__global__ __launch_bounds__(256) void attn_flash(
    const __bf16* __restrict__ qb, const __bf16* __restrict__ kb,
    const __bf16* __restrict__ vtb, __bf16* __restrict__ ob) {
  int qblk = blockIdx.x;                 // 0..7
  int bh   = blockIdx.y;                 // b*16 + h
  int b = bh >> 4, h = bh & 15;
  int tid = threadIdx.x, wave = tid >> 5, lane = tid & 31;
  int lane15 = lane & 15;
  int rsel   = (lane & 16) ? 8 : 0;

  const __bf16* qbase  = qb  + (size_t)bh * 1024 * 64;
  const __bf16* kbase  = kb  + (size_t)bh * 1024 * 64;
  const __bf16* vtbase = vtb + (size_t)bh * 64 * 1024;

  int q0 = qblk * 128 + wave * 16;

  bf16x16 aq[2];
  aq[0] = frag_a(qbase, q0, 64, 0,  lane);
  aq[1] = frag_a(qbase, q0, 64, 32, lane);

  f32x8 accO[4] = {};
  float mrow[8], lrow[8];
#pragma unroll
  for (int j = 0; j < 8; ++j) { mrow[j] = -1e30f; lrow[j] = 0.f; }

  __shared__ __bf16 Pl[8][16 * PLD];
  __bf16* P = &Pl[wave][0];
  const float scale = 0.125f;   // 1/sqrt(64)

  for (int kb0 = 0; kb0 < 1024; kb0 += 64) {
    // S = Q @ K^T for 16 queries x 64 keys
    f32x8 S[4] = {};
#pragma unroll
    for (int c = 0; c < 2; ++c)
#pragma unroll
      for (int nt = 0; nt < 4; ++nt) {
        bf16x16 bk = frag_b(kbase, kb0 + nt * 16, 64, c * 32, lane);
        S[nt] = WMMA_BF16(aq[c], bk, S[nt]);
      }
    // online softmax: row stats via pure-VALU 16-lane butterflies
    float mnew[8], rs[8];
#pragma unroll
    for (int j = 0; j < 8; ++j) {
      float v = fmaxf(fmaxf(S[0][j], S[1][j]), fmaxf(S[2][j], S[3][j])) * scale;
      mnew[j] = fmaxf(mrow[j], redmax16(v));
      float rsum = 0.f;
#pragma unroll
      for (int nt = 0; nt < 4; ++nt) {
        float pv = __expf(S[nt][j] * scale - mnew[j]);
        S[nt][j] = pv;
        rsum += pv;
      }
      rs[j] = redsum16(rsum);
    }
#pragma unroll
    for (int j = 0; j < 8; ++j) {
      float alpha = __expf(mrow[j] - mnew[j]);
      lrow[j] = lrow[j] * alpha + rs[j];
      mrow[j] = mnew[j];
#pragma unroll
      for (int dt = 0; dt < 4; ++dt) accO[dt][j] *= alpha;
    }
    // C-layout -> A-layout via wave-private LDS tile (no barrier needed)
#pragma unroll
    for (int j = 0; j < 8; ++j)
#pragma unroll
      for (int nt = 0; nt < 4; ++nt)
        P[(rsel + j) * PLD + nt * 16 + lane15] = (__bf16)S[nt][j];
    asm volatile("" ::: "memory");   // keep DS store -> DS load program order
    // O += P @ V   (V stored transposed [64][1024] -> NT form)
#pragma unroll
    for (int c = 0; c < 2; ++c) {
      bf16x16 ap = frag_a(P, 0, PLD, c * 32, lane);
#pragma unroll
      for (int dt = 0; dt < 4; ++dt) {
        bf16x16 bv = frag_b(vtbase, dt * 16, 1024, kb0 + c * 32, lane);
        accO[dt] = WMMA_BF16(ap, bv, accO[dt]);
      }
    }
    asm volatile("" ::: "memory");   // WAR: next stores after these loads
  }

  // normalize + scatter to o [B*N][1024] bf16
#pragma unroll
  for (int j = 0; j < 8; ++j) {
    float inv = 1.0f / lrow[j];
    int row = q0 + rsel + j;
#pragma unroll
    for (int dt = 0; dt < 4; ++dt) {
      int col = h * 64 + dt * 16 + lane15;
      ob[((size_t)b * 1024 + row) * 1024 + col] = (__bf16)(accO[dt][j] * inv);
    }
  }
}

// ---------------------------------------------------------------------------
extern "C" void kernel_launch(void* const* d_in, const int* in_sizes, int n_in,
                              void* d_out, int out_size, void* d_ws, size_t ws_size,
                              hipStream_t stream) {
  const float* x      = (const float*)d_in[0];
  const float* qkv_w  = (const float*)d_in[1];
  const float* qkv_b  = (const float*)d_in[2];
  const float* out_w  = (const float*)d_in[3];
  const float* out_b  = (const float*)d_in[4];
  const float* ffn_w1 = (const float*)d_in[5];
  const float* ffn_b1 = (const float*)d_in[6];
  const float* ffn_w2 = (const float*)d_in[7];
  const float* ffn_b2 = (const float*)d_in[8];
  const float* ln1_g  = (const float*)d_in[9];
  const float* ln1_b  = (const float*)d_in[10];
  const float* ln2_g  = (const float*)d_in[11];
  const float* ln2_b  = (const float*)d_in[12];

  const int D = 1024, F = 4096, M = 8192;       // M = B*N tokens

  char* p = (char*)d_ws;
  auto carve = [&](size_t bytes) -> char* {
    char* r = p;
    p += (bytes + 255) & ~(size_t)255;
    return r;
  };
  __bf16* hbf   = (__bf16*)carve((size_t)M * D * 2);       // LN1 out
  __bf16* wqkvT = (__bf16*)carve((size_t)3 * D * D * 2);   // [3D][D]
  __bf16* woutT = (__bf16*)carve((size_t)D * D * 2);       // [D][D]
  __bf16* w1T   = (__bf16*)carve((size_t)F * D * 2);       // [F][D]
  __bf16* w2T   = (__bf16*)carve((size_t)D * F * 2);       // [D][F]
  __bf16* qbuf  = (__bf16*)carve((size_t)M * D * 2);       // [B,H,N,64]
  __bf16* kbuf  = (__bf16*)carve((size_t)M * D * 2);       // [B,H,N,64]
  __bf16* vtbuf = (__bf16*)carve((size_t)M * D * 2);       // [B,H,64,N]
  __bf16* obuf  = (__bf16*)carve((size_t)M * D * 2);       // attn out
  float*  x1    = (float*) carve((size_t)M * D * 4);       // residual 1
  __bf16* h2    = (__bf16*)carve((size_t)M * D * 2);       // LN2 out
  __bf16* ffh   = (__bf16*)carve((size_t)M * F * 2);       // GELU(FFN1)
  float*  y     = (float*)d_out;

  dim3 tb(32, 8);
  // weight convert+transpose passes
  transpose_to_bf16<<<dim3(3 * D / 32, D / 32), tb, 0, stream>>>(qkv_w, wqkvT, D, 3 * D);
  transpose_to_bf16<<<dim3(D / 32, D / 32),     tb, 0, stream>>>(out_w, woutT, D, D);
  transpose_to_bf16<<<dim3(F / 32, D / 32),     tb, 0, stream>>>(ffn_w1, w1T, D, F);
  transpose_to_bf16<<<dim3(D / 32, F / 32),     tb, 0, stream>>>(ffn_w2, w2T, F, D);

  // LN1 -> bf16
  ln_to_bf16<<<M, 256, 0, stream>>>(x, ln1_g, ln1_b, hbf);

  // QKV projection with scatter epilogue
  gemm_nt<EP_QKV><<<dim3(3 * D / 128, M / 128), 256, 0, stream>>>(
      hbf, wqkvT, qkv_b, nullptr, nullptr, nullptr, qbuf, kbuf, vtbuf, M, 3 * D, D);

  // flash attention
  attn_flash<<<dim3(8, 128), 256, 0, stream>>>(qbuf, kbuf, vtbuf, obuf);

  // out projection + residual add (fp32)
  gemm_nt<EP_RESID><<<dim3(D / 128, M / 128), 256, 0, stream>>>(
      obuf, woutT, out_b, nullptr, x1, x, nullptr, nullptr, nullptr, M, D, D);

  // LN2 -> bf16
  ln_to_bf16<<<M, 256, 0, stream>>>(x1, ln2_g, ln2_b, h2);

  // FFN1 + exact GELU
  gemm_nt<EP_GELU><<<dim3(F / 128, M / 128), 256, 0, stream>>>(
      h2, w1T, ffn_b1, ffh, nullptr, nullptr, nullptr, nullptr, nullptr, M, F, D);

  // FFN2 + residual add -> final output (fp32)
  gemm_nt<EP_RESID><<<dim3(D / 128, M / 128), 256, 0, stream>>>(
      ffh, w2T, ffn_b2, nullptr, y, x1, nullptr, nullptr, nullptr, M, D, F);
}
